// HashMLP_14843406975121
// MI455X (gfx1250) — compile-verified, matrix operand order checked by reading, not run on hardware
//
#include <hip/hip_runtime.h>
#include <hip/hip_bf16.h>

// ---------------------------------------------------------------------------
// Fully-fused HashGrid + MLP (24 -> 256 -> 256 -> 256 -> 3) for gfx1250.
// f16 WMMA (v_wmma_f32_16x16x32_f16) with f32 accumulate, weights resident in
// the 320KB WGP LDS, persistent workgroups, 64-point tiles, wave32.
// ---------------------------------------------------------------------------

typedef _Float16 h8  __attribute__((ext_vector_type(8)));
typedef _Float16 h16 __attribute__((ext_vector_type(16)));
typedef float    f8  __attribute__((ext_vector_type(8)));

#define N_POINTS 1048576
#define TILE     64
#define NTILES   (N_POINTS / TILE)
#define HIDDEN   256
#define NLEV     12
#define TSIZE    65536
#define NBLOCKS  512

// tcnn level constants: scale_l = 4*1.61^l - 1 ; res_l = ceil(scale_l)+1
__device__ __constant__ float kScale[NLEV] = {
    3.0f, 5.44f, 9.3684f, 15.693124f, 25.87592964f, 42.2702467204f,
    68.665097219844f, 111.16080652394884f, 179.57889850355763f,
    289.7320265907278f, 467.0785628110717f, 752.6064861258255f};
__device__ __constant__ unsigned kRes[NLEV] = {
    4u, 7u, 11u, 17u, 27u, 44u, 70u, 113u, 181u, 291u, 469u, 754u};

union Frag { h16 v; h8 h[2]; };

__device__ __forceinline__ unsigned umin_u(unsigned a, unsigned b) { return a < b ? a : b; }

// XOR chunk swizzle (chunk = 8 halfs = 16B) so consecutive rows land on
// different LDS bank groups despite the 512B row stride.
__device__ __forceinline__ int idx256(int row, int k) {
  return ((((k >> 3) ^ (row & 15)) << 3) | (k & 7));
}
__device__ __forceinline__ int idx32(int row, int k) {
  return ((((k >> 3) ^ ((row >> 2) & 3)) << 3) | (k & 7));
}

// A/B fragment load, ISA 16-bit 16x32 layout:
// lanes 0-15 -> K chunks {kb*4+0, kb*4+2}; lanes 16-31 -> {kb*4+1, kb*4+3}
__device__ __forceinline__ h16 frag256(const _Float16 (*arr)[HIDDEN], int row, int kb, int sel) {
  const _Float16* rp = arr[row];
  int s  = row & 15;
  int c0 = (kb << 2) + sel;
  int c1 = c0 + 2;
  Frag f;
  f.h[0] = *(const h8*)(rp + ((c0 ^ s) << 3));
  f.h[1] = *(const h8*)(rp + ((c1 ^ s) << 3));
  return f.v;
}
__device__ __forceinline__ h16 frag32(const _Float16 (*arr)[32], int row, int sel) {
  const _Float16* rp = arr[row];
  int s = (row >> 2) & 3;
  Frag f;
  f.h[0] = *(const h8*)(rp + (((0 + sel) ^ s) << 3));
  f.h[1] = *(const h8*)(rp + (((2 + sel) ^ s) << 3));
  return f.v;
}

__global__ __launch_bounds__(256, 1)
void HashMLP_fused_kernel(const float* __restrict__ x,
                          const float* __restrict__ tables,
                          const float* __restrict__ W1, const float* __restrict__ b1,
                          const float* __restrict__ W2, const float* __restrict__ b2,
                          const float* __restrict__ W3, const float* __restrict__ b3,
                          const float* __restrict__ Wo, const float* __restrict__ bo,
                          float* __restrict__ out) {
  // ---- resident LDS: ~319 KB of the 320 KB WGP budget ----
  __shared__ __align__(16) _Float16 sEnc[TILE][32];          //   4 KB
  __shared__ __align__(16) _Float16 sAct[TILE][HIDDEN];      //  32 KB
  __shared__ __align__(16) _Float16 sW1[HIDDEN][32];         //  16 KB
  __shared__ __align__(16) _Float16 sW2[HIDDEN][HIDDEN];     // 128 KB
  __shared__ __align__(16) _Float16 sW3[HIDDEN][HIDDEN];     // 128 KB
  __shared__ __align__(16) _Float16 sWo[16][HIDDEN];         //   8 KB
  __shared__ float sB1[HIDDEN], sB2[HIDDEN], sB3[HIDDEN], sBo[16];

  const int tid  = threadIdx.x;
  const int lane = tid & 31;
  const int wave = tid >> 5;
  const int lm   = lane & 15;          // M (A) / N (B,C) index within half-wave
  const int sel  = (lane >> 4) & 1;    // which K-chunk pair this lane holds
  const int rowbase = (wave >> 1) << 4;  // 4 row-groups of 16 points
  const int ch      = wave & 1;          // column half: 128 cols per wave

  // ---- one-time weight staging: f32 global -> f16 LDS, transposed [n][k] ----
  for (int i = tid; i < HIDDEN * 32; i += 256) {
    int n = i >> 5, k = i & 31;
    sW1[n][idx32(n, k)] = (k < 24) ? (_Float16)W1[k * HIDDEN + n] : (_Float16)0.0f;
  }
  for (int i = tid; i < HIDDEN * HIDDEN; i += 256) {
    int k = i >> 8, n = i & 255;   // coalesced global reads along n
    sW2[n][idx256(n, k)] = (_Float16)W2[k * HIDDEN + n];
    sW3[n][idx256(n, k)] = (_Float16)W3[k * HIDDEN + n];
  }
  for (int i = tid; i < 16 * HIDDEN; i += 256) {
    int k = i >> 4, n = i & 15;
    sWo[n][idx256(n, k)] = (n < 3) ? (_Float16)Wo[k * 3 + n] : (_Float16)0.0f;
  }
  if (tid < HIDDEN) { sB1[tid] = b1[tid]; sB2[tid] = b2[tid]; sB3[tid] = b3[tid]; }
  if (tid < 16) sBo[tid] = (tid < 3) ? bo[tid] : 0.0f;
  __syncthreads();

  const float2* tab2 = (const float2*)tables;

  for (int tile = blockIdx.x; tile < NTILES; tile += gridDim.x) {
    // ================= Phase 1: hash-grid encode (1 pt x 3 levels / thread) ==
    {
      const int p  = tid & 63;
      const int g  = tid >> 6;           // level group 0..3
      const int pg = tile * TILE + p;
      const float xx = x[2 * pg + 0];
      const float yy = x[2 * pg + 1];
      #pragma unroll
      for (int j = 0; j < 3; ++j) {
        const int l = g * 3 + j;
        const float s = kScale[l];
        const unsigned res = kRes[l];
        float px = xx * s + 0.5f, py = yy * s + 0.5f;
        float fx = floorf(px),    fy = floorf(py);
        float wx = px - fx,       wy = py - fy;
        unsigned x0 = (unsigned)fx, y0 = (unsigned)fy;
        unsigned x1 = x0 + 1u,      y1 = y0 + 1u;
        unsigned i00, i10, i01, i11;
        if ((unsigned long long)res * res <= (unsigned long long)TSIZE) {
          unsigned rm = res - 1u;
          unsigned cx0 = umin_u(x0, rm), cx1 = umin_u(x1, rm);
          unsigned cy0 = umin_u(y0, rm), cy1 = umin_u(y1, rm);
          i00 = cx0 + cy0 * res; i10 = cx1 + cy0 * res;
          i01 = cx0 + cy1 * res; i11 = cx1 + cy1 * res;
        } else {
          const unsigned P = 2654435761u;
          i00 = (x0 ^ (y0 * P)) & 65535u; i10 = (x1 ^ (y0 * P)) & 65535u;
          i01 = (x0 ^ (y1 * P)) & 65535u; i11 = (x1 ^ (y1 * P)) & 65535u;
        }
        const float2* tab = tab2 + (size_t)l * TSIZE;
        float2 f00 = tab[i00], f10 = tab[i10], f01 = tab[i01], f11 = tab[i11];
        float w00 = (1.0f - wx) * (1.0f - wy), w10 = wx * (1.0f - wy);
        float w01 = (1.0f - wx) * wy,          w11 = wx * wy;
        float o0 = w00 * f00.x + w10 * f10.x + w01 * f01.x + w11 * f11.x;
        float o1 = w00 * f00.y + w10 * f10.y + w01 * f01.y + w11 * f11.y;
        sEnc[p][idx32(p, 2 * l + 0)] = (_Float16)o0;
        sEnc[p][idx32(p, 2 * l + 1)] = (_Float16)o1;
      }
      if (g == 3) {   // zero-pad K = 24..31
        #pragma unroll
        for (int k = 24; k < 32; ++k) sEnc[p][idx32(p, k)] = (_Float16)0.0f;
      }
    }
    __syncthreads();

    // ================= Layer 1: 24(->32) -> 256, one WMMA per output tile ===
    {
      h16 A = frag32((const _Float16(*)[32])sEnc, rowbase + lm, sel);
      #pragma unroll
      for (int nt = 0; nt < 8; ++nt) {
        const int col = (ch << 7) + (nt << 4) + lm;
        h16 B = frag32((const _Float16(*)[32])sW1, col, sel);
        f8 acc = {};
        acc = __builtin_amdgcn_wmma_f32_16x16x32_f16(false, A, false, B,
                                                     (short)0, acc, false, false);
        const float bb = sB1[col];
        #pragma unroll
        for (int r = 0; r < 8; ++r) {
          const int row = rowbase + r + (sel << 3);
          sAct[row][idx256(row, col)] = (_Float16)fmaxf(acc[r] + bb, 0.0f);
        }
      }
    }
    __syncthreads();

    // ================= Layers 2 & 3: 256 -> 256, K in 8 blocks of 32 ========
    #pragma unroll 1
    for (int layer = 0; layer < 2; ++layer) {
      const _Float16 (*Wt)[HIDDEN] =
          (layer == 0) ? (const _Float16(*)[HIDDEN])sW2 : (const _Float16(*)[HIDDEN])sW3;
      const float* bias = (layer == 0) ? sB2 : sB3;
      f8 acc[8] = {};
      #pragma unroll
      for (int kb = 0; kb < 8; ++kb) {
        h16 A = frag256((const _Float16(*)[HIDDEN])sAct, rowbase + lm, kb, sel);
        #pragma unroll
        for (int nt = 0; nt < 8; ++nt) {
          const int n = (ch << 7) + (nt << 4) + lm;
          h16 B = frag256(Wt, n, kb, sel);
          acc[nt] = __builtin_amdgcn_wmma_f32_16x16x32_f16(false, A, false, B,
                                                           (short)0, acc[nt], false, false);
        }
      }
      __syncthreads();   // everyone done READING sAct before in-place update
      #pragma unroll
      for (int nt = 0; nt < 8; ++nt) {
        const int col = (ch << 7) + (nt << 4) + lm;
        const float bb = bias[col];
        #pragma unroll
        for (int r = 0; r < 8; ++r) {
          const int row = rowbase + r + (sel << 3);
          sAct[row][idx256(row, col)] = (_Float16)fmaxf(acc[nt][r] + bb, 0.0f);
        }
      }
      __syncthreads();
    }

    // ================= Output layer: 256 -> 3 (N padded to 16) ==============
    if (wave < 4) {
      const int rb = wave << 4;
      f8 acc = {};
      #pragma unroll
      for (int kb = 0; kb < 8; ++kb) {
        h16 A = frag256((const _Float16(*)[HIDDEN])sAct, rb + lm, kb, sel);
        h16 B = frag256((const _Float16(*)[HIDDEN])sWo, lm, kb, sel);
        acc = __builtin_amdgcn_wmma_f32_16x16x32_f16(false, A, false, B,
                                                     (short)0, acc, false, false);
      }
      const int col = lm;
      if (col < 3) {
        const float bb = sBo[col];
        #pragma unroll
        for (int r = 0; r < 8; ++r) {
          const int orow = tile * TILE + rb + r + (sel << 3);
          out[orow * 3 + col] = acc[r] + bb;
        }
      }
    }
    // next iteration's post-encode __syncthreads() fences sAct reuse
  }
}

extern "C" void kernel_launch(void* const* d_in, const int* in_sizes, int n_in,
                              void* d_out, int out_size, void* d_ws, size_t ws_size,
                              hipStream_t stream) {
  const float* x      = (const float*)d_in[0];
  const float* tables = (const float*)d_in[1];
  const float* W1     = (const float*)d_in[2];
  const float* b1     = (const float*)d_in[3];
  const float* W2     = (const float*)d_in[4];
  const float* b2     = (const float*)d_in[5];
  const float* W3     = (const float*)d_in[6];
  const float* b3     = (const float*)d_in[7];
  const float* Wo     = (const float*)d_in[8];
  const float* bo     = (const float*)d_in[9];
  float* out = (float*)d_out;
  (void)in_sizes; (void)n_in; (void)out_size; (void)d_ws; (void)ws_size;

  HashMLP_fused_kernel<<<dim3(NBLOCKS), dim3(256), 0, stream>>>(
      x, tables, W1, b1, W2, b2, W3, b3, Wo, bo, out);
}
